// SimVQ_45148696216566
// MI455X (gfx1250) — matrix operand order repaired and checked by reading, not compiled, not run on hardware
//
#include <hip/hip_runtime.h>
#include <hip/hip_bf16.h>

typedef __attribute__((ext_vector_type(16))) __bf16 v16bf;
typedef __attribute__((ext_vector_type(8)))  float  v8f;

#define DIMD 512
#define BM   128
#define BN   64
#define RS   520   // padded LDS row stride in bf16 elements (512 + 8)

__device__ __forceinline__ unsigned short f32_to_bf16(float f) {
  unsigned int u = __float_as_uint(f);
  u += 0x7FFFu + ((u >> 16) & 1u);           // round to nearest even
  return (unsigned short)(u >> 16);
}

union FragU { uint4 q[2]; v16bf v; };

// ---------------- Pass 1: c2[k] = ||codebook_k||^2 (exact f32) ----------------
__global__ __launch_bounds__(128) void k_c2(const float* __restrict__ cb,
                                            float* __restrict__ c2) {
  __shared__ float red[128];
  const int r = blockIdx.x;
  float s = 0.f;
  for (int d = threadIdx.x; d < DIMD; d += 128) {
    float v = cb[(size_t)r * DIMD + d];
    s += v * v;
  }
  red[threadIdx.x] = s; __syncthreads();
  for (int off = 64; off > 0; off >>= 1) {
    if (threadIdx.x < off) red[threadIdx.x] += red[threadIdx.x + off];
    __syncthreads();
  }
  if (threadIdx.x == 0) c2[r] = red[0];
}

// ---------------- Pass 2: fused bf16-WMMA GEMM + argmin ----------------------
__global__ __launch_bounds__(256) void k_argmin(const float* __restrict__ x,
                                                const float* __restrict__ cb,
                                                const float* __restrict__ c2,
                                                int*   __restrict__ idx_out,
                                                float* __restrict__ idxf_out,
                                                int K) {
  __shared__ __align__(16) unsigned short As[BM * RS];
  __shared__ __align__(16) unsigned short Bs[BN * RS];
  __shared__ float c2s[BN];
  __shared__ float rv[BM][2];
  __shared__ int   ri[BM][2];

  const int tid  = threadIdx.x;
  const int lane = tid & 31;
  const int wv   = tid >> 5;      // 8 waves
  const int wm   = wv >> 1;       // 0..3  (M direction, 32 rows each)
  const int wn   = wv & 1;        // 0..1  (N direction, 32 cols each)
  const int l15  = lane & 15;
  const int hi   = (lane >> 4) & 1;
  const int nbase = blockIdx.x * BM;

  // Prefetch first B tile (64 rows * 2KB = 1024 x 128B lines) into WGP cache.
#pragma unroll
  for (int p = 0; p < 4; ++p)
    __builtin_prefetch(cb + ((size_t)tid + p * 256) * 32, 0, 0);

  // Stage A tile (x rows) once, f32 -> bf16.
  for (int it = 0; it < (BM * DIMD / 4) / 256; ++it) {   // 64 iters
    int f  = it * 256 + tid;
    int r  = f >> 7;               // 128 float4 per row
    int d4 = (f & 127) << 2;
    float4 v = *(const float4*)&x[(size_t)(nbase + r) * DIMD + d4];
    ushort4 p;
    p.x = f32_to_bf16(v.x); p.y = f32_to_bf16(v.y);
    p.z = f32_to_bf16(v.z); p.w = f32_to_bf16(v.w);
    *(ushort4*)&As[r * RS + d4] = p;
  }

  float minv[2][8];
  int   mini[2][8];
#pragma unroll
  for (int g = 0; g < 2; ++g)
#pragma unroll
    for (int i = 0; i < 8; ++i) { minv[g][i] = 3.402823466e38f; mini[g][i] = 0; }

  for (int kb = 0; kb < K; kb += BN) {
    __syncthreads();   // previous compute done with Bs (also covers A staging)
    // Stage B tile (codebook rows kb..kb+63), f32 -> bf16.
    for (int it = 0; it < (BN * DIMD / 4) / 256; ++it) {  // 32 iters
      int f  = it * 256 + tid;
      int r  = f >> 7;
      int d4 = (f & 127) << 2;
      float4 v = *(const float4*)&cb[(size_t)(kb + r) * DIMD + d4];
      ushort4 p;
      p.x = f32_to_bf16(v.x); p.y = f32_to_bf16(v.y);
      p.z = f32_to_bf16(v.z); p.w = f32_to_bf16(v.w);
      *(ushort4*)&Bs[r * RS + d4] = p;
    }
    if (tid < BN) c2s[tid] = c2[kb + tid];
    __syncthreads();

    // Prefetch next B tile so its lines sit in WGP$/L2 while we run WMMAs.
    if (kb + BN < K) {
      const float* nxt = cb + (size_t)(kb + BN) * DIMD;
#pragma unroll
      for (int p = 0; p < 4; ++p)
        __builtin_prefetch(nxt + ((size_t)tid + p * 256) * 32, 0, 0);
    }

    v8f acc[2][2];
#pragma unroll
    for (int mt = 0; mt < 2; ++mt)
#pragma unroll
      for (int nt = 0; nt < 2; ++nt)
#pragma unroll
        for (int e = 0; e < 8; ++e) acc[mt][nt][e] = 0.f;

#pragma unroll
    for (int d0 = 0; d0 < DIMD; d0 += 32) {
      // A frags: lanes 0-15 hold K d0..d0+7 / d0+16..23 ; lanes 16-31 shifted by 8
      FragU a0, a1, b0, b1;
      const int aoff0 = (wm * 32 + l15) * RS + d0 + 8 * hi;
      a0.q[0] = *(const uint4*)&As[aoff0];
      a0.q[1] = *(const uint4*)&As[aoff0 + 16];
      const int aoff1 = aoff0 + 16 * RS;
      a1.q[0] = *(const uint4*)&As[aoff1];
      a1.q[1] = *(const uint4*)&As[aoff1 + 16];
      // B frags: same K layout as A (d0..d0+7 / d0+16..23 ; lanes 16-31 shifted by 8)
      const int boff0 = (wn * 32 + l15) * RS + d0 + 8 * hi;
      b0.q[0] = *(const uint4*)&Bs[boff0];
      b0.q[1] = *(const uint4*)&Bs[boff0 + 16];
      const int boff1 = boff0 + 16 * RS;
      b1.q[0] = *(const uint4*)&Bs[boff1];
      b1.q[1] = *(const uint4*)&Bs[boff1 + 16];

      acc[0][0] = __builtin_amdgcn_wmma_f32_16x16x32_bf16(false, a0.v, false, b0.v,
                                                          (short)0, acc[0][0], false, false);
      acc[0][1] = __builtin_amdgcn_wmma_f32_16x16x32_bf16(false, a0.v, false, b1.v,
                                                          (short)0, acc[0][1], false, false);
      acc[1][0] = __builtin_amdgcn_wmma_f32_16x16x32_bf16(false, a1.v, false, b0.v,
                                                          (short)0, acc[1][0], false, false);
      acc[1][1] = __builtin_amdgcn_wmma_f32_16x16x32_bf16(false, a1.v, false, b1.v,
                                                          (short)0, acc[1][1], false, false);
    }

    // Score s = ||c||^2 - 2 x.c ; running argmin per (row, lane) slot.
#pragma unroll
    for (int nt = 0; nt < 2; ++nt) {
      const float c2v = c2s[wn * 32 + nt * 16 + l15];
      const int   col = kb + wn * 32 + nt * 16 + l15;
#pragma unroll
      for (int mt = 0; mt < 2; ++mt)
#pragma unroll
        for (int i = 0; i < 8; ++i) {
          float s = fmaf(-2.f, acc[mt][nt][i], c2v);
          if (s < minv[mt][i]) { minv[mt][i] = s; mini[mt][i] = col; }
        }
    }
  }

  // Reduce across the 16 lanes of each half (same rows, different columns).
#pragma unroll
  for (int off = 1; off <= 8; off <<= 1) {
#pragma unroll
    for (int g = 0; g < 2; ++g)
#pragma unroll
      for (int i = 0; i < 8; ++i) {
        float ov = __shfl_xor(minv[g][i], off, 32);
        int   oi = __shfl_xor(mini[g][i], off, 32);
        if (ov < minv[g][i] || (ov == minv[g][i] && oi < mini[g][i])) {
          minv[g][i] = ov; mini[g][i] = oi;
        }
      }
  }
  if (l15 == 0) {
#pragma unroll
    for (int g = 0; g < 2; ++g)
#pragma unroll
      for (int i = 0; i < 8; ++i) {
        int r = wm * 32 + g * 16 + hi * 8 + i;
        rv[r][wn] = minv[g][i];
        ri[r][wn] = mini[g][i];
      }
  }
  __syncthreads();
  if (tid < BM) {
    float v0 = rv[tid][0], v1 = rv[tid][1];
    int   i0 = ri[tid][0], i1 = ri[tid][1];
    int best = (v1 < v0 || (v1 == v0 && i1 < i0)) ? i1 : i0;
    idx_out[nbase + tid]  = best;
    idxf_out[nbase + tid] = (float)best;
  }
}

// ---------------- Pass 3: gather quantized rows + per-row loss partials ------
__global__ __launch_bounds__(256) void k_gather(const float* __restrict__ x,
                                                const float* __restrict__ cb,
                                                const int*   __restrict__ idx,
                                                float* __restrict__ outq,
                                                float* __restrict__ part) {
  __shared__ float red[256];
  const int n = blockIdx.x;
  const int k = idx[n];
  float s = 0.f;
  for (int d = threadIdx.x; d < DIMD; d += 256) {
    float q  = cb[(size_t)k * DIMD + d];
    float xv = x[(size_t)n * DIMD + d];
    outq[(size_t)n * DIMD + d] = q;   // quantized_st == quantized value-wise
    float df = xv - q;
    s += df * df;
  }
  red[threadIdx.x] = s; __syncthreads();
  for (int off = 128; off > 0; off >>= 1) {
    if (threadIdx.x < off) red[threadIdx.x] += red[threadIdx.x + off];
    __syncthreads();
  }
  if (threadIdx.x == 0) part[n] = red[0];
}

// ---------------- Pass 4: final deterministic loss reduce --------------------
__global__ __launch_bounds__(256) void k_loss(const float* __restrict__ part,
                                              float* __restrict__ out,
                                              int n, float inv) {
  __shared__ float red[256];
  float s = 0.f;
  for (int i = threadIdx.x; i < n; i += 256) s += part[i];
  red[threadIdx.x] = s; __syncthreads();
  for (int off = 128; off > 0; off >>= 1) {
    if (threadIdx.x < off) red[threadIdx.x] += red[threadIdx.x + off];
    __syncthreads();
  }
  if (threadIdx.x == 0) out[0] = red[0] * inv;
}

extern "C" void kernel_launch(void* const* d_in, const int* in_sizes, int n_in,
                              void* d_out, int out_size, void* d_ws, size_t ws_size,
                              hipStream_t stream) {
  (void)n_in; (void)out_size; (void)ws_size;
  const float* x  = (const float*)d_in[0];   // [N, 512] flattened
  const float* cb = (const float*)d_in[1];   // [K, 512]
  const int N = in_sizes[0] / DIMD;          // 8192
  const int K = in_sizes[1] / DIMD;          // 16384

  float* outq = (float*)d_out;               // N*512 quantized
  float* outi = outq + (size_t)N * DIMD;     // N indices (as float)
  float* outl = outi + N;                    // 1 loss scalar

  float* c2   = (float*)d_ws;                // K floats
  int*   idx  = (int*)(c2 + K);              // N ints
  float* part = (float*)(idx + N);           // N floats

  k_c2    <<<K,        128, 0, stream>>>(cb, c2);
  k_argmin<<<N / BM,   256, 0, stream>>>(x, cb, c2, idx, outi, K);
  k_gather<<<N,        256, 0, stream>>>(x, cb, idx, outq, part);
  k_loss  <<<1,        256, 0, stream>>>(part, outl, N, 1.0f / (float)((size_t)N * DIMD));
}
